// EPSparseMoE_51144470561317
// MI455X (gfx1250) — matrix operand order repaired and checked by reading, not compiled, not run on hardware
//
#include <hip/hip_runtime.h>
#include <cstdint>

// ---------- types ----------
typedef __bf16 bf16_t;
typedef __attribute__((ext_vector_type(8)))  __bf16 v8bf;
typedef __attribute__((ext_vector_type(16))) __bf16 v16bf;
typedef __attribute__((ext_vector_type(8)))  float  v8f;

// ---------- constants ----------
constexpr int T_TOK = 4096;   // B*L
constexpr int DIM_D = 1024;
constexpr int DIM_F = 2048;
constexpr int N_EXP = 8;

constexpr int BM = 128;       // token tile
constexpr int BN = 64;        // output-col tile
constexpr int BK = 32;        // k tile == WMMA K for bf16
constexpr int LDS_STRIDE = BK + 8;  // 40 halves = 80B = 5*16B -> every row 16B aligned

// ---------- helpers ----------
__device__ __forceinline__ uint16_t f2bf_bits(float f) {
  uint32_t u = __builtin_bit_cast(uint32_t, f);
  u += 0x7FFFu + ((u >> 16) & 1u);           // round-to-nearest-even
  return (uint16_t)(u >> 16);
}
__device__ __forceinline__ bf16_t f2bf(float f) {
  uint16_t b = f2bf_bits(f);
  return __builtin_bit_cast(bf16_t, b);
}
// packed f32x2 -> bf16x2 (one VALU op); D[15:0]=cvt(a), D[31:16]=cvt(b)
__device__ __forceinline__ uint32_t cvt_pk_bf16(float a, float b) {
  uint32_t r;
  asm("v_cvt_pk_bf16_f32 %0, %1, %2" : "=v"(r) : "v"(a), "v"(b));
  return r;
}
// LDS byte offset of a generic shared pointer (LDS aperture: addr[31:0] == offset)
__device__ __forceinline__ uint32_t lds_off(const void* p) {
  return (uint32_t)(uintptr_t)p;
}
// async 16B copy global -> LDS (GLOBAL_LOAD_ASYNC_TO_LDS_B128, tracked by ASYNCcnt)
__device__ __forceinline__ void async_copy_b128(uint32_t dst_lds, const void* src) {
  asm volatile("global_load_async_to_lds_b128 %0, %1, off"
               :: "v"(dst_lds), "v"((uint64_t)(uintptr_t)src) : "memory");
}
__device__ __forceinline__ void wait_asynccnt0() {
  asm volatile("s_wait_asynccnt 0x0" ::: "memory");
}

// A fragment (16x32 bf16, row-major in LDS): lane holds row (lane&15),
// K = klo..klo+7 (VGPR0-3) and klo+16..klo+23 (VGPR4-7), klo = 8*(lane>>4).
__device__ __forceinline__ v16bf load_frag_a(const bf16_t* sm, int row, int klo) {
  const uint4* p0 = reinterpret_cast<const uint4*>(sm + row * LDS_STRIDE + klo);
  const uint4* p1 = reinterpret_cast<const uint4*>(sm + row * LDS_STRIDE + klo + 16);
  v8bf lo = __builtin_bit_cast(v8bf, *p0);
  v8bf hi = __builtin_bit_cast(v8bf, *p1);
  return __builtin_shufflevector(lo, hi, 0,1,2,3,4,5,6,7,8,9,10,11,12,13,14,15);
}
// B fragment (32x16 bf16): LDS stores B transposed [n][k]; lane holds column
// (lane&15), K = kb..kb+7 and kb+16..kb+23, kb = 8*(lane>>4).
__device__ __forceinline__ v16bf load_frag_b(const bf16_t* sm, int row, int kb) {
  const uint4* p = reinterpret_cast<const uint4*>(sm + row * LDS_STRIDE + kb);
  v8bf lo = __builtin_bit_cast(v8bf, p[0]);
  v8bf hi = __builtin_bit_cast(v8bf, p[2]);
  return __builtin_shufflevector(lo, hi, 0,1,2,3,4,5,6,7,8,9,10,11,12,13,14,15);
}

// ---------- shared double-buffered GEMM mainloop ----------
// C[BMxBN] (f32 acc) = A(bf16, lda) x Bg(f32->bf16, ldb), K-reduction.
// A rows optionally gathered via token list; gather hoisted out of the k-loop.
__device__ __forceinline__ void moe_gemm_mainloop(
    const bf16_t* __restrict__ A, int lda,
    const float* __restrict__ Bg, int ldb,
    int K, int mb, int n0,
    const int* __restrict__ gather, int Te,
    bf16_t* sA, bf16_t* sB,   // each sized 2 * tile (double buffered)
    v8f (&acc)[2][2])
{
  const int tid  = threadIdx.x;
  const int lane = tid & 31;
  const int wave = tid >> 5;
  const int wm = wave & 3;     // 4 waves along M (32 rows each)
  const int wn = wave >> 2;    // 2 waves along N (32 cols each)

  // -- loop-invariant staging coordinates --
  const int r = tid >> 1, s = tid & 1;          // A: (row, 32B half-row)
  const int kr = tid >> 3, cg = (tid & 7) * 8;  // B: (k-row, 8-col group)
  const bf16_t* arow;                            // gathered A row base (hoisted)
  if (gather) {
    const int p = mb + r;
    const int t = (p < Te) ? gather[p] : gather[0];
    arow = A + (size_t)t * lda;
  } else {
    arow = A + (size_t)(mb + r) * lda;
  }
  const int aBuf = BM * LDS_STRIDE;  // buffer stride (halves)
  const int bBuf = BN * LDS_STRIDE;

  // stage one k-tile into buffer `buf`
  auto stage = [&](int k0, int buf) {
    // A tile: pure bf16 copy -> async DMA to LDS, 2 x 16B per thread
    const uint32_t dst = lds_off(sA + buf * aBuf + r * LDS_STRIDE + s * 16);
    async_copy_b128(dst,      arow + k0 + s * 16);
    async_copy_b128(dst + 16, arow + k0 + s * 16 + 8);
    // B tile: f32 load, packed cvt to bf16, transposed store [n][k]
    const float4* g =
        reinterpret_cast<const float4*>(Bg + (size_t)(k0 + kr) * ldb + n0 + cg);
    float4 f0 = g[0];
    float4 f1 = g[1];
    uint32_t p01 = cvt_pk_bf16(f0.x, f0.y);
    uint32_t p23 = cvt_pk_bf16(f0.z, f0.w);
    uint32_t p45 = cvt_pk_bf16(f1.x, f1.y);
    uint32_t p67 = cvt_pk_bf16(f1.z, f1.w);
    uint16_t* sb = reinterpret_cast<uint16_t*>(sB + buf * bBuf);
    sb[(cg + 0) * LDS_STRIDE + kr] = (uint16_t)p01;
    sb[(cg + 1) * LDS_STRIDE + kr] = (uint16_t)(p01 >> 16);
    sb[(cg + 2) * LDS_STRIDE + kr] = (uint16_t)p23;
    sb[(cg + 3) * LDS_STRIDE + kr] = (uint16_t)(p23 >> 16);
    sb[(cg + 4) * LDS_STRIDE + kr] = (uint16_t)p45;
    sb[(cg + 5) * LDS_STRIDE + kr] = (uint16_t)(p45 >> 16);
    sb[(cg + 6) * LDS_STRIDE + kr] = (uint16_t)p67;
    sb[(cg + 7) * LDS_STRIDE + kr] = (uint16_t)(p67 >> 16);
  };

  // prologue: fill buffer 0
  stage(0, 0);
  wait_asynccnt0();
  __syncthreads();

  const int arf = lane & 15;
  const int klo = (lane >> 4) * 8;
  const int kb  = (lane >> 4) * 8;

  for (int k0 = 0; k0 < K; k0 += BK) {
    const int cur = (k0 / BK) & 1;
    const int nxt = cur ^ 1;
    if (k0 + BK < K) stage(k0 + BK, nxt);   // overlap with WMMA below

    const bf16_t* cA = sA + cur * aBuf;
    const bf16_t* cB = sB + cur * bBuf;
    v16bf a0 = load_frag_a(cA, wm * 32 +      arf, klo);
    v16bf a1 = load_frag_a(cA, wm * 32 + 16 + arf, klo);
    v16bf b0 = load_frag_b(cB, wn * 32 +      arf, kb);
    v16bf b1 = load_frag_b(cB, wn * 32 + 16 + arf, kb);

    acc[0][0] = __builtin_amdgcn_wmma_f32_16x16x32_bf16(false, a0, false, b0,
                                                        (short)0, acc[0][0], false, false);
    acc[0][1] = __builtin_amdgcn_wmma_f32_16x16x32_bf16(false, a0, false, b1,
                                                        (short)0, acc[0][1], false, false);
    acc[1][0] = __builtin_amdgcn_wmma_f32_16x16x32_bf16(false, a1, false, b0,
                                                        (short)0, acc[1][0], false, false);
    acc[1][1] = __builtin_amdgcn_wmma_f32_16x16x32_bf16(false, a1, false, b1,
                                                        (short)0, acc[1][1], false, false);

    wait_asynccnt0();       // next-buffer async A writes landed
    __syncthreads();        // compiler adds s_wait_dscnt for the B ds_stores
  }
}

// ---------- kernel 0: zero output region + routing counters ----------
__global__ __launch_bounds__(256) void moe_zero_kernel(float4* out4, int n4, int* cnt) {
  const int id = blockIdx.x * 256 + threadIdx.x;
  if (id < n4) out4[id] = make_float4(0.f, 0.f, 0.f, 0.f);
  if (blockIdx.x == 0 && threadIdx.x < N_EXP) cnt[threadIdx.x] = 0;
}

// ---------- kernel 1: X f32 -> bf16 (packed convert) ----------
__global__ __launch_bounds__(256) void moe_cvt_x_kernel(const float* __restrict__ x,
                                                        uint2* __restrict__ xb, int n4) {
  const int id = blockIdx.x * 256 + threadIdx.x;
  if (id >= n4) return;
  float4 v = reinterpret_cast<const float4*>(x)[id];
  xb[id] = make_uint2(cvt_pk_bf16(v.x, v.y), cvt_pk_bf16(v.z, v.w));
}

// ---------- kernel 2: router (one wave32 per token) ----------
__global__ __launch_bounds__(256) void moe_router_kernel(
    const float* __restrict__ x, const float* __restrict__ Wg,
    const float* __restrict__ bg, float* __restrict__ logits,
    float* __restrict__ cw, int* __restrict__ cnt, int* __restrict__ tlist)
{
  const int lane  = threadIdx.x & 31;
  const int token = blockIdx.x * 8 + (threadIdx.x >> 5);
  const float* xr = x + (size_t)token * DIM_D;

  float acc[N_EXP] = {0.f, 0.f, 0.f, 0.f, 0.f, 0.f, 0.f, 0.f};
  for (int d = lane; d < DIM_D; d += 32) {
    const float xv = xr[d];
    const float4* w = reinterpret_cast<const float4*>(Wg + (size_t)d * N_EXP);
    float4 w0 = w[0], w1 = w[1];
    acc[0] += xv * w0.x; acc[1] += xv * w0.y; acc[2] += xv * w0.z; acc[3] += xv * w0.w;
    acc[4] += xv * w1.x; acc[5] += xv * w1.y; acc[6] += xv * w1.z; acc[7] += xv * w1.w;
  }
#pragma unroll
  for (int e = 0; e < N_EXP; ++e)
#pragma unroll
    for (int off = 16; off > 0; off >>= 1) acc[e] += __shfl_xor(acc[e], off, 32);

  if (lane == 0) {
    float l[N_EXP];
#pragma unroll
    for (int e = 0; e < N_EXP; ++e) {
      l[e] = acc[e] + bg[e];
      logits[token * N_EXP + e] = l[e];
    }
    int i1 = 0;
#pragma unroll
    for (int e = 1; e < N_EXP; ++e) if (l[e] > l[i1]) i1 = e;
    int i2 = (i1 == 0) ? 1 : 0;
#pragma unroll
    for (int e = 0; e < N_EXP; ++e) if (e != i1 && l[e] > l[i2]) i2 = e;
    const float e2  = __expf(l[i2] - l[i1]);     // softmax over top-2, max-shifted
    const float inv = 1.0f / (1.0f + e2);
#pragma unroll
    for (int e = 0; e < N_EXP; ++e) cw[token * N_EXP + e] = 0.0f;
    cw[token * N_EXP + i1] = inv;
    cw[token * N_EXP + i2] = e2 * inv;
    // compacted per-expert token lists (order nondeterministic; output isn't)
    int p1 = atomicAdd(&cnt[i1], 1); tlist[i1 * T_TOK + p1] = token;
    int p2 = atomicAdd(&cnt[i2], 1); tlist[i2 * T_TOK + p2] = token;
  }
}

// ---------- kernel 3: gathered GEMM1 + bias + SiLU -> H (bf16) ----------
__global__ __launch_bounds__(256) void moe_gemm1_silu_kernel(
    const bf16_t* __restrict__ Xb, const float* __restrict__ W1e,
    const float* __restrict__ b1e, const int* __restrict__ tlist_e,
    const int* __restrict__ cnt, int e, bf16_t* __restrict__ H)
{
  const int Te = cnt[e];
  const int mb = blockIdx.y * BM;
  if (mb >= Te) return;                       // sparse early-exit
  const int n0 = blockIdx.x * BN;

  __shared__ __align__(16) bf16_t sA[2 * BM * LDS_STRIDE];
  __shared__ __align__(16) bf16_t sB[2 * BN * LDS_STRIDE];

  v8f acc[2][2] = {};
  moe_gemm_mainloop(Xb, DIM_D, W1e, DIM_F, DIM_D, mb, n0, tlist_e, Te, sA, sB, acc);

  const int lane = threadIdx.x & 31;
  const int wave = threadIdx.x >> 5;
  const int wm = wave & 3, wn = wave >> 2;
#pragma unroll
  for (int i = 0; i < 2; ++i)
#pragma unroll
    for (int j = 0; j < 2; ++j) {
      const int nG = n0 + wn * 32 + j * 16 + (lane & 15);
      const int mG = mb + wm * 32 + i * 16 + ((lane >> 4) * 8);
      const float bias = b1e[nG];
#pragma unroll
      for (int r = 0; r < 8; ++r) {
        float v = acc[i][j][r] + bias;
        v = v * (1.0f / (1.0f + __expf(-v)));              // SiLU
        H[(size_t)(mG + r) * DIM_F + nG] = f2bf(v);        // position-indexed rows
      }
    }
}

// ---------- kernel 4: GEMM2 + bias, weighted scatter-accumulate to out ----------
__global__ __launch_bounds__(256) void moe_gemm2_scatter_kernel(
    const bf16_t* __restrict__ H, const float* __restrict__ W2e,
    const float* __restrict__ b2e, const float* __restrict__ cw,
    const int* __restrict__ tlist_e, const int* __restrict__ cnt, int e,
    float* __restrict__ out)
{
  const int Te = cnt[e];
  const int mb = blockIdx.y * BM;
  if (mb >= Te) return;
  const int n0 = blockIdx.x * BN;

  __shared__ __align__(16) bf16_t sA[2 * BM * LDS_STRIDE];
  __shared__ __align__(16) bf16_t sB[2 * BN * LDS_STRIDE];

  v8f acc[2][2] = {};
  moe_gemm_mainloop(H, DIM_F, W2e, DIM_D, DIM_F, mb, n0, nullptr, 0, sA, sB, acc);

  const int lane = threadIdx.x & 31;
  const int wave = threadIdx.x >> 5;
  const int wm = wave & 3, wn = wave >> 2;
#pragma unroll
  for (int i = 0; i < 2; ++i)
#pragma unroll
    for (int j = 0; j < 2; ++j) {
      const int nG = n0 + wn * 32 + j * 16 + (lane & 15);
      const int mG = mb + wm * 32 + i * 16 + ((lane >> 4) * 8);
      const float bias = b2e[nG];
#pragma unroll
      for (int r = 0; r < 8; ++r) {
        const int p = mG + r;
        if (p < Te) {
          const int t = tlist_e[p];
          const float w = cw[t * N_EXP + e];
          out[(size_t)t * DIM_D + nG] += w * (acc[i][j][r] + bias);
        }
      }
    }
}

// ---------- host ----------
extern "C" void kernel_launch(void* const* d_in, const int* in_sizes, int n_in,
                              void* d_out, int out_size, void* d_ws, size_t ws_size,
                              hipStream_t stream) {
  (void)in_sizes; (void)n_in; (void)out_size; (void)ws_size;
  const float* x  = (const float*)d_in[0];   // [T, D]
  const float* Wg = (const float*)d_in[1];   // [D, E]
  const float* bg = (const float*)d_in[2];   // [E]
  const float* W1 = (const float*)d_in[3];   // [E, D, F]
  const float* b1 = (const float*)d_in[4];   // [E, F]
  const float* W2 = (const float*)d_in[5];   // [E, F, D]
  const float* b2 = (const float*)d_in[6];   // [E, D]
  // d_in[7] = top_k (fixed at 2)

  float* out    = (float*)d_out;                       // [T, D]
  float* logits = out + (size_t)T_TOK * DIM_D;         // [T, E]

  // workspace layout
  char* ws = (char*)d_ws;
  bf16_t* Xb   = (bf16_t*)ws;                                         // 8 MiB
  float*  cw   = (float*)(ws + (size_t)8 * 1024 * 1024);              // 128 KiB
  int*    tlst = (int*)  (ws + (size_t)8 * 1024 * 1024 + 128 * 1024); // 128 KiB
  int*    cnt  = (int*)  (ws + (size_t)8 * 1024 * 1024 + 256 * 1024); // 32 B
  bf16_t* H    = (bf16_t*)(ws + (size_t)8 * 1024 * 1024 + 512 * 1024);// 16 MiB

  const int nOut4 = T_TOK * DIM_D / 4;
  moe_zero_kernel<<<(nOut4 + 255) / 256, 256, 0, stream>>>((float4*)out, nOut4, cnt);
  moe_cvt_x_kernel<<<(T_TOK * DIM_D / 4 + 255) / 256, 256, 0, stream>>>(
      x, (uint2*)Xb, T_TOK * DIM_D / 4);
  moe_router_kernel<<<T_TOK / 8, 256, 0, stream>>>(x, Wg, bg, logits, cw, cnt, tlst);

  for (int e = 0; e < N_EXP; ++e) {
    const float* W1e = W1 + (size_t)e * DIM_D * DIM_F;
    const float* b1e = b1 + (size_t)e * DIM_F;
    const float* W2e = W2 + (size_t)e * DIM_F * DIM_D;
    const float* b2e = b2 + (size_t)e * DIM_D;
    const int* te = tlst + (size_t)e * T_TOK;
    moe_gemm1_silu_kernel<<<dim3(DIM_F / BN, T_TOK / BM), 256, 0, stream>>>(
        Xb, W1e, b1e, te, cnt, e, H);
    moe_gemm2_scatter_kernel<<<dim3(DIM_D / BN, T_TOK / BM), 256, 0, stream>>>(
        H, W2e, b2e, cw, te, cnt, e, out);
  }
}
